// DepthCorr_74414603370675
// MI455X (gfx1250) — matrix operand order repaired and checked by reading, not compiled, not run on hardware
//
#include <hip/hip_runtime.h>
#include <hip/hip_bf16.h>

typedef __attribute__((ext_vector_type(16))) _Float16 v16h;
typedef __attribute__((ext_vector_type(8)))  _Float16 v8h;
typedef __attribute__((ext_vector_type(8)))  float    v8f;

#define CIN  256
#define HMID 256
#define KDIM (CIN * 9)     // 2304, ordered as tap*256 + channel
#define NKT  (KDIM / 32)   // 72 k-steps for the 3x3 convs

// ---------------------------------------------------------------------------
// BN folding: scale = gamma * rsqrt(var + eps), bias = beta - mean * scale
// ---------------------------------------------------------------------------
__global__ void bnprep_kernel(const float* __restrict__ g, const float* __restrict__ be,
                              const float* __restrict__ me, const float* __restrict__ va,
                              float* __restrict__ sc, float* __restrict__ bi, int n) {
    int i = blockIdx.x * blockDim.x + threadIdx.x;
    if (i < n) {
        float inv = g[i] * rsqrtf(va[i] + 1e-5f);
        sc[i] = inv;
        bi[i] = be[i] - me[i] * inv;
    }
}

// ---------------------------------------------------------------------------
// NCHW f32 -> NHWC f16, LDS-tiled transpose (32 channels x 32 positions).
// src: [b][256][HW] f32 ; dst: [b][HW][256] f16
// ---------------------------------------------------------------------------
__global__ void __launch_bounds__(256)
cvt_transpose_kernel(const float* __restrict__ src, _Float16* __restrict__ dst, int HW) {
    __shared__ _Float16 tile[32][33];
    const int b  = blockIdx.z;
    const int c0 = blockIdx.y * 32;
    const int p0 = blockIdx.x * 32;
    const int tx = threadIdx.x, ty = threadIdx.y;   // (32, 8)
#pragma unroll
    for (int r = 0; r < 32; r += 8) {
        int c = c0 + ty + r, p = p0 + tx;
        float v = (p < HW) ? src[((size_t)b * 256 + c) * HW + p] : 0.f;
        tile[ty + r][tx] = (_Float16)v;
    }
    __syncthreads();
#pragma unroll
    for (int r = 0; r < 32; r += 8) {
        int p = p0 + ty + r, c = c0 + tx;
        if (p < HW) dst[((size_t)b * HW + p) * 256 + c] = tile[tx][ty + r];
    }
}

// ---------------------------------------------------------------------------
// Pack OIHW f32 weights into per-lane WMMA A fragments (contiguous k order, same as the B loads),
// K ordered as tap*C + c to match NHWC activations.
//   dst[((mt*nkt + kt)*32 + lane)*16 + h]
// lane: g = lane>>4, m = lane&15 ; half h: v=h>>1, j=h&1,
//   Koff = 16*g + h
// ---------------------------------------------------------------------------
__global__ void pack_w_kernel(const float* __restrict__ src, _Float16* __restrict__ dst,
                              int M, int K, int taps) {
    int idx = blockIdx.x * blockDim.x + threadIdx.x;
    if (idx >= M * K) return;
    int h    = idx & 15;
    int lane = (idx >> 4) & 31;
    int tile = idx >> 9;               // mt*nkt + kt
    int nkt  = K >> 5;
    int kt   = tile % nkt;
    int mt   = tile / nkt;
    int g    = lane >> 4;
    int v    = h >> 1, j = h & 1;
    int koff = g * 16 + h;
    int k    = kt * 32 + koff;
    int m    = mt * 16 + (lane & 15);
    float val;
    if (taps == 9) {
        int t = k >> 8;                // tap index (K = t*256 + c)
        int c = k & 255;
        val = src[(m * CIN + c) * 9 + t];
    } else {
        val = src[m * CIN + k];
    }
    dst[idx] = (_Float16)val;
}

// ---------------------------------------------------------------------------
// Implicit-GEMM 3x3 conv (valid, stride 1) + fused BN + ReLU.
// NHWC f16 in / NHWC f16 out.
// One wave computes 64 out-channels x 32 positions:
//   4 M-tiles x 2 N-tiles -> 8 WMMAs per k-step from 4 A loads + 2 B loads
//   (1.5 b128 loads per WMMA).
// ---------------------------------------------------------------------------
__global__ void __launch_bounds__(32)
conv3x3_wmma_kernel(const _Float16* __restrict__ x16,   // NHWC [B][IH][IW][C]
                    const _Float16* __restrict__ wpk,   // packed A fragments
                    const float* __restrict__ sc, const float* __restrict__ bi,
                    _Float16* __restrict__ out16,       // NHWC [B][OH*OW][HMID]
                    int IH, int IW, int OH, int OW) {
    const int lane = threadIdx.x;
    const int nt = blockIdx.x, mq = blockIdx.y, b = blockIdx.z;
    const int npos = OH * OW;
    const int g    = lane >> 4;

    const int n0   = nt * 32 + (lane & 15);
    const int n1   = n0 + 16;
    const int pos0 = (n0 < npos) ? n0 : (npos - 1);    // clamp keeps EXEC full
    const int pos1 = (n1 < npos) ? n1 : (npos - 1);
    const int oy0  = pos0 / OW, ox0 = pos0 - (pos0 / OW) * OW;
    const int oy1  = pos1 / OW, ox1 = pos1 - (pos1 / OW) * OW;

    // per-lane pixel bases; +g*16 selects this lane's 16-channel K slice
    const _Float16* xb0 = x16 + (((size_t)b * IH + oy0) * IW + ox0) * CIN + g * 16;
    const _Float16* xb1 = x16 + (((size_t)b * IH + oy1) * IW + ox1) * CIN + g * 16;
    const _Float16* wp  = wpk + (size_t)(mq * 4) * NKT * 512 + lane * 16;

    v8f acc00 = {}, acc10 = {}, acc20 = {}, acc30 = {};   // N-tile 0
    v8f acc01 = {}, acc11 = {}, acc21 = {}, acc31 = {};   // N-tile 1

    for (int t = 0; t < 9; ++t) {                   // tap loop (uniform)
        const int ky = t / 3, kx = t - (t / 3) * 3;
        const _Float16* xt0 = xb0 + (ky * IW + kx) * CIN;
        const _Float16* xt1 = xb1 + (ky * IW + kx) * CIN;
#pragma unroll
        for (int kc = 0; kc < 8; ++kc) {            // 8 channel-blocks of 32
            const int kt = t * 8 + kc;
            v16h b0 = *(const v16h*)(xt0 + kc * 32);
            v16h b1 = *(const v16h*)(xt1 + kc * 32);
            const _Float16* wpt = wp + (size_t)kt * 512;
            v16h a0 = *(const v16h*)(wpt);
            v16h a1 = *(const v16h*)(wpt + (size_t)1 * NKT * 512);
            v16h a2 = *(const v16h*)(wpt + (size_t)2 * NKT * 512);
            v16h a3 = *(const v16h*)(wpt + (size_t)3 * NKT * 512);
            acc00 = __builtin_amdgcn_wmma_f32_16x16x32_f16(false, a0, false, b0, (short)0, acc00, false, false);
            acc01 = __builtin_amdgcn_wmma_f32_16x16x32_f16(false, a0, false, b1, (short)0, acc01, false, false);
            acc10 = __builtin_amdgcn_wmma_f32_16x16x32_f16(false, a1, false, b0, (short)0, acc10, false, false);
            acc11 = __builtin_amdgcn_wmma_f32_16x16x32_f16(false, a1, false, b1, (short)0, acc11, false, false);
            acc20 = __builtin_amdgcn_wmma_f32_16x16x32_f16(false, a2, false, b0, (short)0, acc20, false, false);
            acc21 = __builtin_amdgcn_wmma_f32_16x16x32_f16(false, a2, false, b1, (short)0, acc21, false, false);
            acc30 = __builtin_amdgcn_wmma_f32_16x16x32_f16(false, a3, false, b0, (short)0, acc30, false, false);
            acc31 = __builtin_amdgcn_wmma_f32_16x16x32_f16(false, a3, false, b1, (short)0, acc31, false, false);
        }
    }

    // epilogue: per M-tile, 8 consecutive channels per lane -> one b128 store
    const int chb = mq * 64 + g * 8;
    _Float16* ob0 = out16 + ((size_t)b * npos + n0) * HMID + chb;
    _Float16* ob1 = out16 + ((size_t)b * npos + n1) * HMID + chb;
    v8f accsN0[4] = {acc00, acc10, acc20, acc30};
    v8f accsN1[4] = {acc01, acc11, acc21, acc31};
#pragma unroll
    for (int i = 0; i < 4; ++i) {
        int ch0 = chb + i * 16;
        v8h o0, o1;
#pragma unroll
        for (int rr = 0; rr < 8; ++rr) {
            float s = sc[ch0 + rr], bb = bi[ch0 + rr];
            float v0 = accsN0[i][rr] * s + bb;
            float v1 = accsN1[i][rr] * s + bb;
            o0[rr] = (_Float16)(v0 > 0.f ? v0 : 0.f);
            o1[rr] = (_Float16)(v1 > 0.f ? v1 : 0.f);
        }
        if (n0 < npos) *(v8h*)(ob0 + i * 16) = o0;
        if (n1 < npos) *(v8h*)(ob1 + i * 16) = o1;
    }
}

// ---------------------------------------------------------------------------
// Depthwise 5x5 xcorr + ReLU, NHWC (fully coalesced: thread = channel).
// sf: [b][29][29][256], kf: [b][5][5][256], corr: [b][625][256]
// One block per (b, output row): 25-tap kernel kept in registers.
// ---------------------------------------------------------------------------
__global__ void __launch_bounds__(256)
dwxcorr_kernel(const _Float16* __restrict__ sf, const _Float16* __restrict__ kf,
               _Float16* __restrict__ corr) {
    const int b = blockIdx.y;
    const int y = blockIdx.x;          // output row 0..24
    const int c = threadIdx.x;         // channel 0..255
    float kk[25];
#pragma unroll
    for (int t = 0; t < 25; ++t)
        kk[t] = (float)kf[((size_t)b * 25 + t) * 256 + c];
    const _Float16* srow = sf + (((size_t)b * 29 + y) * 29) * 256 + c;
    _Float16* orow = corr + ((size_t)b * 625 + y * 25) * 256 + c;
    for (int x = 0; x < 25; ++x) {
        float acc = 0.f;
#pragma unroll
        for (int i = 0; i < 5; ++i)
#pragma unroll
            for (int j = 0; j < 5; ++j)
                acc += (float)srow[((size_t)i * 29 + x + j) * 256] * kk[i * 5 + j];
        orow[(size_t)x * 256] = (_Float16)(acc > 0.f ? acc : 0.f);
    }
}

// ---------------------------------------------------------------------------
// 1x1 conv WMMA GEMM: M=256 out ch, K=256 mid ch (NHWC corr -> contiguous
// B fragments), 4 M-tiles x 2 N-tiles per wave, fused BN + ReLU,
// NCHW f32 output (reference layout).
// ---------------------------------------------------------------------------
__global__ void __launch_bounds__(32)
conv1x1_wmma_kernel(const _Float16* __restrict__ corr,  // NHWC [b][625][256]
                    const _Float16* __restrict__ wpk,
                    const float* __restrict__ sc, const float* __restrict__ bi,
                    float* __restrict__ out) {
    const int lane = threadIdx.x;
    const int nt = blockIdx.x, mq = blockIdx.y, b = blockIdx.z;
    const int npos = 625;
    const int g    = lane >> 4;

    const int n0   = nt * 32 + (lane & 15);
    const int n1   = n0 + 16;
    const int pos0 = (n0 < npos) ? n0 : (npos - 1);
    const int pos1 = (n1 < npos) ? n1 : (npos - 1);

    const _Float16* cb0 = corr + ((size_t)b * npos + pos0) * HMID + g * 16;
    const _Float16* cb1 = corr + ((size_t)b * npos + pos1) * HMID + g * 16;
    const _Float16* wp  = wpk + (size_t)(mq * 4) * 8 * 512 + lane * 16;

    v8f acc00 = {}, acc10 = {}, acc20 = {}, acc30 = {};
    v8f acc01 = {}, acc11 = {}, acc21 = {}, acc31 = {};
#pragma unroll
    for (int kt = 0; kt < 8; ++kt) {
        v16h b0 = *(const v16h*)(cb0 + kt * 32);
        v16h b1 = *(const v16h*)(cb1 + kt * 32);
        const _Float16* wpt = wp + (size_t)kt * 512;
        v16h a0 = *(const v16h*)(wpt);
        v16h a1 = *(const v16h*)(wpt + (size_t)1 * 8 * 512);
        v16h a2 = *(const v16h*)(wpt + (size_t)2 * 8 * 512);
        v16h a3 = *(const v16h*)(wpt + (size_t)3 * 8 * 512);
        acc00 = __builtin_amdgcn_wmma_f32_16x16x32_f16(false, a0, false, b0, (short)0, acc00, false, false);
        acc01 = __builtin_amdgcn_wmma_f32_16x16x32_f16(false, a0, false, b1, (short)0, acc01, false, false);
        acc10 = __builtin_amdgcn_wmma_f32_16x16x32_f16(false, a1, false, b0, (short)0, acc10, false, false);
        acc11 = __builtin_amdgcn_wmma_f32_16x16x32_f16(false, a1, false, b1, (short)0, acc11, false, false);
        acc20 = __builtin_amdgcn_wmma_f32_16x16x32_f16(false, a2, false, b0, (short)0, acc20, false, false);
        acc21 = __builtin_amdgcn_wmma_f32_16x16x32_f16(false, a2, false, b1, (short)0, acc21, false, false);
        acc30 = __builtin_amdgcn_wmma_f32_16x16x32_f16(false, a3, false, b0, (short)0, acc30, false, false);
        acc31 = __builtin_amdgcn_wmma_f32_16x16x32_f16(false, a3, false, b1, (short)0, acc31, false, false);
    }

    v8f accsN0[4] = {acc00, acc10, acc20, acc30};
    v8f accsN1[4] = {acc01, acc11, acc21, acc31};
#pragma unroll
    for (int i = 0; i < 4; ++i) {
        int ch0 = mq * 64 + i * 16 + g * 8;
#pragma unroll
        for (int rr = 0; rr < 8; ++rr) {
            float s = sc[ch0 + rr], bb = bi[ch0 + rr];
            float v0 = accsN0[i][rr] * s + bb;
            float v1 = accsN1[i][rr] * s + bb;
            v0 = v0 > 0.f ? v0 : 0.f;
            v1 = v1 > 0.f ? v1 : 0.f;
            size_t rowbase = ((size_t)b * CIN + ch0 + rr) * npos;
            if (n0 < npos) out[rowbase + n0] = v0;
            if (n1 < npos) out[rowbase + n1] = v1;
        }
    }
}

// ---------------------------------------------------------------------------
extern "C" void kernel_launch(void* const* d_in, const int* in_sizes, int n_in,
                              void* d_out, int out_size, void* d_ws, size_t ws_size,
                              hipStream_t stream) {
    const float* kernel_in = (const float*)d_in[0];
    const float* search_in = (const float*)d_in[1];
    const float* Wk = (const float*)d_in[2];
    const float* gk = (const float*)d_in[3];
    const float* bk = (const float*)d_in[4];
    const float* mk = (const float*)d_in[5];
    const float* vk = (const float*)d_in[6];
    const float* Ws = (const float*)d_in[7];
    const float* gs = (const float*)d_in[8];
    const float* bs = (const float*)d_in[9];
    const float* ms = (const float*)d_in[10];
    const float* vs = (const float*)d_in[11];
    const float* Wf = (const float*)d_in[12];
    const float* gf = (const float*)d_in[13];
    const float* bf = (const float*)d_in[14];
    const float* mf = (const float*)d_in[15];
    const float* vf = (const float*)d_in[16];
    float* out = (float*)d_out;

    const int Bn = 128;

    size_t off = 0;
    auto alloc = [&](size_t bytes) -> char* {
        char* p = (char*)d_ws + off;
        off = (off + bytes + 255) & ~(size_t)255;
        return p;
    };
    _Float16* k16  = (_Float16*)alloc((size_t)Bn * 49 * 256 * 2);    // NHWC kernel input
    _Float16* s16  = (_Float16*)alloc((size_t)Bn * 961 * 256 * 2);   // NHWC search input
    _Float16* wpkK = (_Float16*)alloc((size_t)256 * KDIM * 2);
    _Float16* wpkS = (_Float16*)alloc((size_t)256 * KDIM * 2);
    _Float16* wpkF = (_Float16*)alloc((size_t)256 * 256 * 2);
    _Float16* kf16 = (_Float16*)alloc((size_t)Bn * 25 * 256 * 2);    // NHWC [b][5*5][256]
    _Float16* sf16 = (_Float16*)alloc((size_t)Bn * 841 * 256 * 2);   // NHWC [b][29*29][256]
    float* scK = (float*)alloc(256 * 4); float* biK = (float*)alloc(256 * 4);
    float* scS = (float*)alloc(256 * 4); float* biS = (float*)alloc(256 * 4);
    float* scF = (float*)alloc(256 * 4); float* biF = (float*)alloc(256 * 4);
    // corr (41 MB, NHWC [b][625][256]) aliases the dead search-f16 buffer (63 MB)
    _Float16* corr16 = s16;

    bnprep_kernel<<<1, 256, 0, stream>>>(gk, bk, mk, vk, scK, biK, 256);
    bnprep_kernel<<<1, 256, 0, stream>>>(gs, bs, ms, vs, scS, biS, 256);
    bnprep_kernel<<<1, 256, 0, stream>>>(gf, bf, mf, vf, scF, biF, 256);

    // NCHW f32 -> NHWC f16 transposes
    cvt_transpose_kernel<<<dim3(2, 8, Bn),  dim3(32, 8), 0, stream>>>(kernel_in, k16, 49);
    cvt_transpose_kernel<<<dim3(31, 8, Bn), dim3(32, 8), 0, stream>>>(search_in, s16, 961);

    pack_w_kernel<<<(256 * KDIM + 255) / 256, 256, 0, stream>>>(Wk, wpkK, 256, KDIM, 9);
    pack_w_kernel<<<(256 * KDIM + 255) / 256, 256, 0, stream>>>(Ws, wpkS, 256, KDIM, 9);
    pack_w_kernel<<<(256 * 256 + 255) / 256, 256, 0, stream>>>(Wf, wpkF, 256, 256, 1);

    // kernel branch: 7x7 -> 5x5  (25 positions -> 1 N-tile of 32, 4 M-quads)
    conv3x3_wmma_kernel<<<dim3(1, 4, Bn), 32, 0, stream>>>(k16, wpkK, scK, biK, kf16, 7, 7, 5, 5);
    // search branch: 31x31 -> 29x29  (841 positions -> 27 N-tiles of 32)
    conv3x3_wmma_kernel<<<dim3(27, 4, Bn), 32, 0, stream>>>(s16, wpkS, scS, biS, sf16, 31, 31, 29, 29);

    // depthwise 5x5 xcorr + ReLU (coalesced NHWC)
    dwxcorr_kernel<<<dim3(25, Bn), 256, 0, stream>>>(sf16, kf16, corr16);

    // 1x1 conv + BN + ReLU -> NCHW f32 output (625 positions -> 20 N-tiles of 32)
    conv1x1_wmma_kernel<<<dim3(20, 4, Bn), 32, 0, stream>>>(corr16, wpkF, scF, biF, out);
}
